// AlignIF_61314953117885
// MI455X (gfx1250) — matrix operand order repaired and checked by reading, not compiled, not run on hardware
//
#include <hip/hip_runtime.h>
#include <math.h>

typedef __attribute__((ext_vector_type(16))) __bf16 v16bf;
typedef __attribute__((ext_vector_type(8)))  __bf16 v8bf;
typedef __attribute__((ext_vector_type(8)))  float  v8f;

#define HID 128

// ---------------- small math helpers ----------------
struct F3 { float x, y, z; };
__device__ __forceinline__ F3 f3sub(F3 a, F3 b) { return {a.x - b.x, a.y - b.y, a.z - b.z}; }
__device__ __forceinline__ float f3dot(F3 a, F3 b) { return a.x * b.x + a.y * b.y + a.z * b.z; }
__device__ __forceinline__ F3 f3cross(F3 a, F3 b) {
  return {a.y * b.z - a.z * b.y, a.z * b.x - a.x * b.z, a.x * b.y - a.y * b.x};
}
__device__ __forceinline__ F3 f3norm(F3 a) {
  float n = fmaxf(sqrtf(f3dot(a, a)), 1e-12f);
  return {a.x / n, a.y / n, a.z / n};
}
__device__ __forceinline__ float fsign(float v) { return (v > 0.f) ? 1.f : ((v < 0.f) ? -1.f : 0.f); }

__device__ __forceinline__ v8f zero8() {
  v8f z;
#pragma unroll
  for (int i = 0; i < 8; i++) z[i] = 0.f;
  return z;
}

// Build WMMA bf16 A fragment from row-major LDS row: elements 0..15 = K[kb..kb+15]
// (contiguous k order, matching the B fragment loads).
__device__ __forceinline__ v16bf load_a16(const __bf16* p) {
  v8bf lo = *(const v8bf*)p;
  v8bf hi = *(const v8bf*)(p + 8);
  v16bf a;
#pragma unroll
  for (int i = 0; i < 8; i++) { a[i] = lo[i]; a[i + 8] = hi[i]; }
  return a;
}

__device__ __forceinline__ float block_sum_128(float v, volatile float* red, int c) {
  red[c] = v; __syncthreads();
  if (c < 64) red[c] += red[c + 64]; __syncthreads();
  if (c < 32) red[c] += red[c + 32]; __syncthreads();
  if (c < 16) red[c] += red[c + 16]; __syncthreads();
  if (c < 8)  red[c] += red[c + 8];  __syncthreads();
  if (c < 4)  red[c] += red[c + 4];  __syncthreads();
  if (c < 2)  red[c] += red[c + 2];  __syncthreads();
  if (c < 1)  red[c] += red[c + 1];  __syncthreads();
  float r = red[0]; __syncthreads();
  return r;
}

// ---------------- utility kernels ----------------
__global__ void AlignIF_fill_f32(float* __restrict__ p, float v, int n) {
  int i = blockIdx.x * blockDim.x + threadIdx.x;
  if (i < n) p[i] = v;
}

__global__ void AlignIF_count(const int* __restrict__ tgt, float* __restrict__ cnt, int E) {
  int e = blockIdx.x * blockDim.x + threadIdx.x;
  if (e < E) unsafeAtomicAdd(&cnt[tgt[e]], 1.f);
}

// W [K,Ncol] f32 row-major -> Wt [Ncol][K] bf16 (column-major), WMMA B-friendly
__global__ void AlignIF_transpose_w(const float* __restrict__ W, __bf16* __restrict__ Wt,
                                    int K, int Ncol) {
  int idx = blockIdx.x * blockDim.x + threadIdx.x;
  if (idx >= K * Ncol) return;
  int k = idx / Ncol, c = idx % Ncol;
  Wt[(size_t)c * K + k] = (__bf16)W[idx];
}

// ---------------- featurization ----------------
__global__ void AlignIF_feat_node(const float* __restrict__ X, const int* __restrict__ batch,
                                  float* __restrict__ hV4, float* __restrict__ Q,
                                  int* __restrict__ qvalid, int n) {
  int i = blockIdx.x * blockDim.x + threadIdx.x;
  if (i >= n) return;
  auto cs = [&](int k) -> bool { return (k > 0 && k < n) ? (batch[k] != batch[k - 1]) : false; };
  auto getX = [&](int k) -> F3 { return {X[k * 3 + 0], X[k * 3 + 1], X[k * 3 + 2]}; };
  auto getU = [&](int k) -> F3 { return f3norm(f3sub(getX(k + 1), getX(k))); };  // k in [0,n-2]

  // dihedral
  float dih = 0.f;
  if (i >= 1 && i <= n - 3) {
    F3 um1 = getU(i - 1), u0 = getU(i), up1 = getU(i + 1);
    F3 c1 = f3norm(f3cross(um1, u0));
    F3 c2 = f3norm(f3cross(u0, up1));
    float cd = fminf(fmaxf(f3dot(c1, c2), -1.f + 1e-6f), 1.f - 1e-6f);
    dih = acosf(cd) * fsign(f3dot(c2, um1));
  }
  bool invd = cs(i) || cs(i + 1) || cs(i + 2) || (i == 0) || (i == n - 2) || (i == n - 1);
  float dsin = invd ? 0.f : sinf(dih);
  float dcos = invd ? 0.f : cosf(dih);

  // bond angle
  bool interior = (i >= 1 && i <= n - 2);
  float ca = 0.f, sa = 0.f;
  if (interior) {
    F3 x0 = getX(i - 1), x1 = getX(i), x2 = getX(i + 1);
    F3 d0 = f3norm(f3sub(x0, x1));
    F3 d1 = f3norm(f3sub(x2, x1));
    ca = f3dot(d0, d1);
    sa = sqrtf(1.f - ca * ca + 1e-6f);
  }
  bool inva = cs(i) || cs(i + 1) || (i == 0) || (i == n - 1);
  hV4[i * 4 + 0] = dsin;
  hV4[i * 4 + 1] = dcos;
  hV4[i * 4 + 2] = inva ? 0.f : sa;
  hV4[i * 4 + 3] = inva ? 0.f : ca;

  // local frame Q (columns: b, nrm, b x nrm)
  float q[9] = {0, 0, 0, 0, 0, 0, 0, 0, 0};
  if (interior) {
    F3 u0 = getU(i - 1), u1 = getU(i);
    F3 b = f3norm(f3sub(u0, u1));
    F3 nr = f3norm(f3cross(u0, u1));
    F3 bx = f3cross(b, nr);
    q[0] = b.x; q[1] = nr.x; q[2] = bx.x;
    q[3] = b.y; q[4] = nr.y; q[5] = bx.y;
    q[6] = b.z; q[7] = nr.z; q[8] = bx.z;
  }
#pragma unroll
  for (int k = 0; k < 9; k++) Q[i * 9 + k] = q[k];
  qvalid[i] = inva ? 0 : 1;
}

// node embed: hV4[4] @ W[4,128] + b, layernorm; one block (128 thr) per node
__global__ void AlignIF_node_embed(const float* __restrict__ hV4,
                                   const float* __restrict__ W, const float* __restrict__ b,
                                   const float* __restrict__ g, const float* __restrict__ bn,
                                   float* __restrict__ hV, __bf16* __restrict__ hVbf) {
  int i = blockIdx.x, c = threadIdx.x;
  float v = b[c];
#pragma unroll
  for (int k = 0; k < 4; k++) v += hV4[i * 4 + k] * W[k * HID + c];
  __shared__ float red[128];
  float mu = block_sum_128(v, red, c) * (1.f / HID);
  float var = block_sum_128(v * v, red, c) * (1.f / HID) - mu * mu;
  float rs = rsqrtf(var + 1e-5f);
  float o = (v - mu) * rs * g[c] + bn[c];
  hV[(size_t)i * HID + c] = o;
  hVbf[(size_t)i * HID + c] = (__bf16)o;
}

// edge features (quaternion, rbf, direction) + edge embed + layernorm; block(128)/edge
__global__ void AlignIF_edge_embed(const float* __restrict__ X, const float* __restrict__ Q,
                                   const int* __restrict__ qvalid,
                                   const int* __restrict__ src, const int* __restrict__ tgt,
                                   const float* __restrict__ We, const float* __restrict__ be,
                                   const float* __restrict__ g, const float* __restrict__ bn,
                                   __bf16* __restrict__ hEbf) {
  int e = blockIdx.x, c = threadIdx.x;
  __shared__ float raw[23];
  if (c == 0) {
    int s = src[e], t = tgt[e];
    float Qs[9], Qt[9];
#pragma unroll
    for (int k = 0; k < 9; k++) { Qs[k] = Q[s * 9 + k]; Qt[k] = Q[t * 9 + k]; }
    float R[3][3];
#pragma unroll
    for (int j = 0; j < 3; j++)
#pragma unroll
      for (int k = 0; k < 3; k++) {
        float a = 0.f;
#pragma unroll
        for (int ii = 0; ii < 3; ii++) a += Qt[ii * 3 + j] * Qs[ii * 3 + k];
        R[j][k] = a;
      }
    float Rxx = R[0][0], Ryy = R[1][1], Rzz = R[2][2];
    float mx = 0.5f * sqrtf(fabsf(1.f + Rxx - Ryy - Rzz));
    float my = 0.5f * sqrtf(fabsf(1.f - Rxx + Ryy - Rzz));
    float mz = 0.5f * sqrtf(fabsf(1.f - Rxx - Ryy + Rzz));
    float qx = fsign(R[2][1] - R[1][2]) * mx;
    float qy = fsign(R[0][2] - R[2][0]) * my;
    float qz = fsign(R[1][0] - R[0][1]) * mz;
    float w = sqrtf(fmaxf(1.f + Rxx + Ryy + Rzz, 0.f)) * 0.5f;
    float nrm = fmaxf(sqrtf(qx * qx + qy * qy + qz * qz + w * w), 1e-12f);
    float ov = (qvalid[s] && qvalid[t]) ? 1.f : 0.f;
    raw[0] = qx / nrm * ov; raw[1] = qy / nrm * ov; raw[2] = qz / nrm * ov; raw[3] = w / nrm * ov;
    F3 xs = {X[s * 3], X[s * 3 + 1], X[s * 3 + 2]};
    F3 xt = {X[t * 3], X[t * 3 + 1], X[t * 3 + 2]};
    F3 dv = f3sub(xs, xt);
    float D = sqrtf(f3dot(dv, dv) + 1e-6f);
#pragma unroll
    for (int j = 0; j < 16; j++) {
      float muj = j * (20.f / 15.f);
      float z = (D - muj) / 1.25f;
      raw[4 + j] = expf(-z * z);
    }
    F3 dn = f3norm(dv);
    float tv = qvalid[t] ? 1.f : 0.f;
#pragma unroll
    for (int j = 0; j < 3; j++)
      raw[20 + j] = (Qt[0 * 3 + j] * dn.x + Qt[1 * 3 + j] * dn.y + Qt[2 * 3 + j] * dn.z) * tv;
  }
  __syncthreads();
  float v = be[c];
#pragma unroll
  for (int k = 0; k < 23; k++) v += raw[k] * We[k * HID + c];
  __shared__ float red[128];
  float mu = block_sum_128(v, red, c) * (1.f / HID);
  float var = block_sum_128(v * v, red, c) * (1.f / HID) - mu * mu;
  float rs = rsqrtf(var + 1e-5f);
  hEbf[(size_t)e * HID + c] = (__bf16)((v - mu) * rs * g[c] + bn[c]);
}

// ---------------- edge message kernel: 16 edges x 128 cols per block ----------------
__global__ __launch_bounds__(256) void AlignIF_msg(
    const __bf16* __restrict__ hVbf, const __bf16* __restrict__ hEbf,
    const int* __restrict__ src, const int* __restrict__ tgt,
    const __bf16* __restrict__ W1t, const float* __restrict__ b1,
    const __bf16* __restrict__ W2t, const float* __restrict__ b2,
    const __bf16* __restrict__ W3t, const float* __restrict__ b3,
    float* __restrict__ agg) {
  __shared__ alignas(16) __bf16 sA[16][384];
  __shared__ alignas(16) __bf16 sM[16][128];
  __shared__ alignas(16) __bf16 sM2[16][128];
  __shared__ int sTgt[16];

  const int tid = threadIdx.x;
  {
    const int el = tid >> 4, part = tid & 15;
    const int e = blockIdx.x * 16 + el;
    const int s = src[e], t = tgt[e];
    if (part == 0) sTgt[el] = t;
#pragma unroll
    for (int j = 0; j < 3; j++) {
      int off = part * 24 + j * 8;  // 8-elem chunks never cross the 128/256 boundaries
      const __bf16* sp;
      if (off < 128)      sp = hVbf + (size_t)t * HID + off;
      else if (off < 256) sp = hEbf + (size_t)e * HID + (off - 128);
      else                sp = hVbf + (size_t)s * HID + (off - 256);
      *(v8bf*)&sA[el][off] = *(const v8bf*)sp;
    }
  }
  __syncthreads();

  const int lane = tid & 31, wave = tid >> 5;
  const int arow = lane & 15, ak = (lane >> 4) * 16;
  const int bk = (lane >> 4) * 16;
  const int m0 = (lane >> 4) * 8;
  const int ccol = wave * 16 + (lane & 15);

  // GEMM1: [16,384] x [384,16] per wave, 12 k-steps
  v8f acc = zero8();
#pragma unroll
  for (int ks = 0; ks < 12; ks++) {
    int kb = ks * 32;
    v16bf a = load_a16(&sA[arow][kb + ak]);
    v16bf b = *(const v16bf*)(W1t + (size_t)ccol * 384 + kb + bk);
    acc = __builtin_amdgcn_wmma_f32_16x16x32_bf16(false, a, false, b, (short)0, acc, false, false);
  }
  {
    float bias = b1[ccol];
#pragma unroll
    for (int r = 0; r < 8; r++) sM[m0 + r][ccol] = (__bf16)fmaxf(acc[r] + bias, 0.f);
  }
  __syncthreads();

  // GEMM2: K=128
  acc = zero8();
#pragma unroll
  for (int ks = 0; ks < 4; ks++) {
    int kb = ks * 32;
    v16bf a = load_a16(&sM[arow][kb + ak]);
    v16bf b = *(const v16bf*)(W2t + (size_t)ccol * 128 + kb + bk);
    acc = __builtin_amdgcn_wmma_f32_16x16x32_bf16(false, a, false, b, (short)0, acc, false, false);
  }
  {
    float bias = b2[ccol];
#pragma unroll
    for (int r = 0; r < 8; r++) sM2[m0 + r][ccol] = (__bf16)fmaxf(acc[r] + bias, 0.f);
  }
  __syncthreads();

  // GEMM3: K=128, scatter-add to agg[tgt]
  acc = zero8();
#pragma unroll
  for (int ks = 0; ks < 4; ks++) {
    int kb = ks * 32;
    v16bf a = load_a16(&sM2[arow][kb + ak]);
    v16bf b = *(const v16bf*)(W3t + (size_t)ccol * 128 + kb + bk);
    acc = __builtin_amdgcn_wmma_f32_16x16x32_bf16(false, a, false, b, (short)0, acc, false, false);
  }
  {
    float bias = b3[ccol];
#pragma unroll
    for (int r = 0; r < 8; r++)
      unsafeAtomicAdd(&agg[(size_t)sTgt[m0 + r] * HID + ccol], acc[r] + bias);
  }
}

// ---------------- node update: mean-agg + LN1 + FFN (WMMA) + LN2 ----------------
__global__ __launch_bounds__(256) void AlignIF_node_update(
    float* __restrict__ hV, __bf16* __restrict__ hVbf,
    const float* __restrict__ agg, const float* __restrict__ cnt,
    const float* __restrict__ g1, const float* __restrict__ bb1,
    const __bf16* __restrict__ F1t, const float* __restrict__ fb1,
    const __bf16* __restrict__ F2t, const float* __restrict__ fb2,
    const float* __restrict__ g2, const float* __restrict__ bb2) {
  __shared__ alignas(16) __bf16 sX[16][128];
  __shared__ float sXf[16][128];
  __shared__ alignas(16) __bf16 sH[16][512];
  __shared__ float sY[16][128];
  __shared__ float redA[16][16];
  __shared__ float redB[16][16];
  __shared__ float stat[16][2];

  const int tid = threadIdx.x;
  const int row = tid >> 4, part = tid & 15;
  const int node = blockIdx.x * 16 + row;

  float xr[8];
  {
    float inv = 1.f / fmaxf(cnt[node], 1.f);
    float ps = 0.f, pq = 0.f;
#pragma unroll
    for (int j = 0; j < 8; j++) {
      int col = part * 8 + j;
      float v = hV[(size_t)node * HID + col] + agg[(size_t)node * HID + col] * inv;
      xr[j] = v; ps += v; pq += v * v;
    }
    redA[row][part] = ps; redB[row][part] = pq;
  }
  __syncthreads();
  if (part == 0) {
    float s = 0.f, q = 0.f;
    for (int k = 0; k < 16; k++) { s += redA[row][k]; q += redB[row][k]; }
    float mu = s * (1.f / HID);
    stat[row][0] = mu;
    stat[row][1] = rsqrtf(q * (1.f / HID) - mu * mu + 1e-5f);
  }
  __syncthreads();
  {
    float mu = stat[row][0], rs = stat[row][1];
#pragma unroll
    for (int j = 0; j < 8; j++) {
      int col = part * 8 + j;
      float v = (xr[j] - mu) * rs * g1[col] + bb1[col];
      sXf[row][col] = v;
      sX[row][col] = (__bf16)v;
    }
  }
  __syncthreads();

  const int lane = tid & 31, wave = tid >> 5;
  const int arow = lane & 15, ak = (lane >> 4) * 16;
  const int bk = (lane >> 4) * 16;
  const int m0 = (lane >> 4) * 8;

  // FF1: 128 -> 512 (each wave covers 64 cols = 4 tiles)
  for (int ct = 0; ct < 4; ct++) {
    int bcol = wave * 64 + ct * 16 + (lane & 15);
    v8f acc = zero8();
#pragma unroll
    for (int ks = 0; ks < 4; ks++) {
      int kb = ks * 32;
      v16bf a = load_a16(&sX[arow][kb + ak]);
      v16bf b = *(const v16bf*)(F1t + (size_t)bcol * 128 + kb + bk);
      acc = __builtin_amdgcn_wmma_f32_16x16x32_bf16(false, a, false, b, (short)0, acc, false, false);
    }
    float bias = fb1[bcol];
#pragma unroll
    for (int r = 0; r < 8; r++) sH[m0 + r][bcol] = (__bf16)fmaxf(acc[r] + bias, 0.f);
  }
  __syncthreads();

  // FF2: 512 -> 128 (16 k-steps) + residual
  {
    int bcol = wave * 16 + (lane & 15);
    v8f acc = zero8();
#pragma unroll
    for (int ks = 0; ks < 16; ks++) {
      int kb = ks * 32;
      v16bf a = load_a16(&sH[arow][kb + ak]);
      v16bf b = *(const v16bf*)(F2t + (size_t)bcol * 512 + kb + bk);
      acc = __builtin_amdgcn_wmma_f32_16x16x32_bf16(false, a, false, b, (short)0, acc, false, false);
    }
    float bias = fb2[bcol];
#pragma unroll
    for (int r = 0; r < 8; r++) {
      int m = m0 + r;
      sY[m][bcol] = sXf[m][bcol] + acc[r] + bias;
    }
  }
  __syncthreads();

  // LN2 + writeback (f32 + bf16)
  {
    float ps = 0.f, pq = 0.f;
#pragma unroll
    for (int j = 0; j < 8; j++) {
      int col = part * 8 + j;
      float v = sY[row][col];
      xr[j] = v; ps += v; pq += v * v;
    }
    redA[row][part] = ps; redB[row][part] = pq;
  }
  __syncthreads();
  if (part == 0) {
    float s = 0.f, q = 0.f;
    for (int k = 0; k < 16; k++) { s += redA[row][k]; q += redB[row][k]; }
    float mu = s * (1.f / HID);
    stat[row][0] = mu;
    stat[row][1] = rsqrtf(q * (1.f / HID) - mu * mu + 1e-5f);
  }
  __syncthreads();
  {
    float mu = stat[row][0], rs = stat[row][1];
#pragma unroll
    for (int j = 0; j < 8; j++) {
      int col = part * 8 + j;
      float v = (xr[j] - mu) * rs * g2[col] + bb2[col];
      hV[(size_t)node * HID + col] = v;
      hVbf[(size_t)node * HID + col] = (__bf16)v;
    }
  }
}

// ---------------- output head ----------------
__global__ void AlignIF_out(const float* __restrict__ hV, const float* __restrict__ W,
                            const float* __restrict__ b, float* __restrict__ out, int n) {
  int idx = blockIdx.x * blockDim.x + threadIdx.x;
  if (idx >= n * 4) return;
  int i = idx >> 2, j = idx & 3;
  float a = b[j];
#pragma unroll 8
  for (int k = 0; k < HID; k++) a += hV[(size_t)i * HID + k] * W[k * 4 + j];
  out[idx] = a;
}

// ---------------- host launcher ----------------
extern "C" void kernel_launch(void* const* d_in, const int* in_sizes, int n_in,
                              void* d_out, int out_size, void* d_ws, size_t ws_size,
                              hipStream_t stream) {
  (void)n_in; (void)out_size; (void)ws_size;
  const int N = in_sizes[1];
  const int E = in_sizes[2] / 2;

  const float* X = (const float*)d_in[0];
  const int* batch = (const int*)d_in[1];
  const int* src = (const int*)d_in[2];
  const int* tgt = src + E;
  auto P = [&](int idx) -> const float* { return (const float*)d_in[idx]; };

  // workspace layout (256B aligned slabs)
  char* base = (char*)d_ws;
  size_t off = 0;
  auto alloc = [&](size_t bytes) -> void* {
    void* p = base + off;
    off = (off + bytes + 255) & ~(size_t)255;
    return p;
  };
  __bf16* hEbf = (__bf16*)alloc((size_t)E * HID * 2);
  float*  hV   = (float*)alloc((size_t)N * HID * 4);
  __bf16* hVbf = (__bf16*)alloc((size_t)N * HID * 2);
  float*  agg  = (float*)alloc((size_t)N * HID * 4);
  float*  cnt  = (float*)alloc((size_t)N * 4);
  float*  Q    = (float*)alloc((size_t)N * 9 * 4);
  int*    qval = (int*)alloc((size_t)N * 4);
  float*  hV4  = (float*)alloc((size_t)N * 4 * 4);
  __bf16 *w1t[6], *w2t[6], *w3t[6], *f1t[6], *f2t[6];
  for (int l = 0; l < 6; l++) {
    w1t[l] = (__bf16*)alloc((size_t)384 * 128 * 2);
    w2t[l] = (__bf16*)alloc((size_t)128 * 128 * 2);
    w3t[l] = (__bf16*)alloc((size_t)128 * 128 * 2);
    f1t[l] = (__bf16*)alloc((size_t)128 * 512 * 2);
    f2t[l] = (__bf16*)alloc((size_t)512 * 128 * 2);
  }

  // in-degree counts (constant across layers)
  AlignIF_fill_f32<<<(N + 255) / 256, 256, 0, stream>>>(cnt, 0.f, N);
  AlignIF_count<<<(E + 255) / 256, 256, 0, stream>>>(tgt, cnt, E);

  // featurize + embeddings
  AlignIF_feat_node<<<(N + 255) / 256, 256, 0, stream>>>(X, batch, hV4, Q, qval, N);
  AlignIF_node_embed<<<N, 128, 0, stream>>>(hV4, P(3), P(4), P(5), P(6), hV, hVbf);
  AlignIF_edge_embed<<<E, 128, 0, stream>>>(X, Q, qval, src, tgt, P(7), P(8), P(9), P(10), hEbf);

  // weight transposes (bf16, column-major for WMMA B fragments)
  for (int l = 0; l < 6; l++) {
    int b0 = 11 + l * 14;
    AlignIF_transpose_w<<<(384 * 128 + 255) / 256, 256, 0, stream>>>(P(b0 + 0), w1t[l], 384, 128);
    AlignIF_transpose_w<<<(128 * 128 + 255) / 256, 256, 0, stream>>>(P(b0 + 2), w2t[l], 128, 128);
    AlignIF_transpose_w<<<(128 * 128 + 255) / 256, 256, 0, stream>>>(P(b0 + 4), w3t[l], 128, 128);
    AlignIF_transpose_w<<<(128 * 512 + 255) / 256, 256, 0, stream>>>(P(b0 + 6), f1t[l], 128, 512);
    AlignIF_transpose_w<<<(512 * 128 + 255) / 256, 256, 0, stream>>>(P(b0 + 8), f2t[l], 512, 128);
  }

  // 6 MPNN layers (3 enc + 3 dec, identical structure)
  for (int l = 0; l < 6; l++) {
    int b0 = 11 + l * 14;
    AlignIF_fill_f32<<<(N * HID + 255) / 256, 256, 0, stream>>>(agg, 0.f, N * HID);
    AlignIF_msg<<<E / 16, 256, 0, stream>>>(hVbf, hEbf, src, tgt,
                                            w1t[l], P(b0 + 1), w2t[l], P(b0 + 3),
                                            w3t[l], P(b0 + 5), agg);
    AlignIF_node_update<<<N / 16, 256, 0, stream>>>(hV, hVbf, agg, cnt,
                                                    P(b0 + 10), P(b0 + 11),
                                                    f1t[l], P(b0 + 7),
                                                    f2t[l], P(b0 + 9),
                                                    P(b0 + 12), P(b0 + 13));
  }

  // output head
  AlignIF_out<<<(N * 4 + 255) / 256, 256, 0, stream>>>(hV, P(95), P(96), (float*)d_out, N);
}
